// ContextualCouplingTSA_72438918414621
// MI455X (gfx1250) — matrix operand order repaired and checked by reading, not compiled, not run on hardware
//
#include <hip/hip_runtime.h>
#include <math.h>

typedef _Float16 half_t;
typedef __attribute__((ext_vector_type(16))) _Float16 v16h;
typedef __attribute__((ext_vector_type(8)))  float    v8f;
typedef int vi4 __attribute__((vector_size(16)));

#define BATCH   4
#define SEQ     2048
#define DMODEL  1024
#define DCH     512
#define NHEADS  8
#define DHEAD   64
#define ROWS    (BATCH * SEQ)          // 8192
#define BH      (BATCH * NHEADS)       // 32

#if __has_builtin(__builtin_amdgcn_global_load_async_to_lds_b128) && \
    __has_builtin(__builtin_amdgcn_s_wait_asynccnt)
#define HAVE_ASYNC_LDS 1
#else
#define HAVE_ASYNC_LDS 0
#endif

// ---------------------------------------------------------------------------
// WMMA tile loaders (documented CDNA5 wave32 layouts)
// A 16x32 f16: lanes 0-15 -> M=lane, K base 0; lanes 16-31 -> K base 8.
//   v<4 : K = kb + 2v + {0,1};  v>=4 : K = kb + 16 + 2(v-4) + {0,1}
// B 32x16 f16: lane%16 = N; same K order as A (K base 8*(lane/16), j>=8 -> +16).
// C/D 16x16 f32: N = lane%16; VGPR r -> M = r + 8*(lane/16)
// ---------------------------------------------------------------------------
__device__ __forceinline__ v16h load_a16x32(const half_t* __restrict__ rowBase) {
    v16h a;
#pragma unroll
    for (int v = 0; v < 8; ++v) {
        const int k = (v < 4) ? (2 * v) : (16 + 2 * (v - 4));
        a[2 * v]     = rowBase[k];
        a[2 * v + 1] = rowBase[k + 1];
    }
    return a;
}

__device__ __forceinline__ v16h load_b32x16_rowmajorN(const half_t* __restrict__ rowBase) {
    v16h b;
#pragma unroll
    for (int j = 0; j < 16; ++j) b[j] = rowBase[(j < 8) ? j : (j + 8)];
    return b;
}

__device__ __forceinline__ v8f wmma_f16(v16h a, v16h b, v8f c) {
    return __builtin_amdgcn_wmma_f32_16x16x32_f16(false, a, false, b, (short)0, c,
                                                  false, false);
}

// ---------------------------------------------------------------------------
// 1. LayerNorm + split
// ---------------------------------------------------------------------------
__global__ void ln_split_kernel(const float* __restrict__ src,
                                const float* __restrict__ g,
                                const float* __restrict__ b,
                                half_t* __restrict__ x1h,
                                float* __restrict__ x2f) {
    const int row = blockIdx.x;
    const float* x = src + (size_t)row * DMODEL;
    __shared__ float sh_s[256], sh_q[256];
    float s = 0.f, q = 0.f;
    for (int i = threadIdx.x; i < DMODEL; i += 256) {
        float v = x[i]; s += v; q += v * v;
    }
    sh_s[threadIdx.x] = s; sh_q[threadIdx.x] = q;
    __syncthreads();
    for (int off = 128; off > 0; off >>= 1) {
        if (threadIdx.x < off) {
            sh_s[threadIdx.x] += sh_s[threadIdx.x + off];
            sh_q[threadIdx.x] += sh_q[threadIdx.x + off];
        }
        __syncthreads();
    }
    const float mean = sh_s[0] * (1.f / DMODEL);
    const float var  = sh_q[0] * (1.f / DMODEL) - mean * mean;
    const float inv  = rsqrtf(var + 1e-5f);
    for (int i = threadIdx.x; i < DMODEL; i += 256) {
        float v = (x[i] - mean) * inv * g[i] + b[i];
        if (i < DCH) x1h[(size_t)row * DCH + i]         = (half_t)v;
        else         x2f[(size_t)row * DCH + (i - DCH)] = v;
    }
}

// ---------------------------------------------------------------------------
// 2. fp32 -> fp16 conversion
// ---------------------------------------------------------------------------
__global__ void cvt_f32_f16_kernel(const float* __restrict__ src,
                                   half_t* __restrict__ dst, int n) {
    int i = blockIdx.x * 256 + threadIdx.x;
    if (i < n) dst[i] = (half_t)src[i];
}

// ---------------------------------------------------------------------------
// 3. Generic GEMM:  C[m,n] = sum_k A[m,k] * W[n,k] + bias[n]
// ---------------------------------------------------------------------------
template <bool HALF_OUT>
__global__ void gemm_wmma_kernel(const half_t* __restrict__ A,
                                 const half_t* __restrict__ W,
                                 const float* __restrict__ bias,
                                 void* __restrict__ Cout,
                                 int M, int N, int Kd) {
    const int wave = threadIdx.x >> 5;
    const int lane = threadIdx.x & 31;
    const int wm = wave >> 1;
    const int wn = wave & 1;
    const int mBase = blockIdx.y * 128 + wm * 32;
    const int nBase = blockIdx.x * 64 + wn * 32;

    const int mrow = mBase + (lane & 15);
    const int ncol = nBase + (lane & 15);
    const int kbA = (lane >> 4) << 3;
    const int kbB = (lane >> 4) << 3;

    v8f acc[2][2] = {};
    for (int k0 = 0; k0 < Kd; k0 += 32) {
        v16h a[2], b[2];
#pragma unroll
        for (int t = 0; t < 2; ++t) {
            const half_t* ra = A + (size_t)(mrow + 16 * t) * Kd + k0 + kbA;
            __builtin_prefetch(ra + 64, 0, 1);
            a[t] = load_a16x32(ra);
        }
#pragma unroll
        for (int t = 0; t < 2; ++t)
            b[t] = load_b32x16_rowmajorN(W + (size_t)(ncol + 16 * t) * Kd + k0 + kbB);
#pragma unroll
        for (int i = 0; i < 2; ++i)
#pragma unroll
            for (int j = 0; j < 2; ++j)
                acc[i][j] = wmma_f16(a[i], b[j], acc[i][j]);
    }
    const int mo = (lane >> 4) * 8;
#pragma unroll
    for (int i = 0; i < 2; ++i)
#pragma unroll
        for (int j = 0; j < 2; ++j) {
            const int nn = nBase + 16 * j + (lane & 15);
            const float bv = bias[nn];
#pragma unroll
            for (int r = 0; r < 8; ++r) {
                const int mm = mBase + 16 * i + mo + r;
                const float val = acc[i][j][r] + bv;
                if (HALF_OUT) ((half_t*)Cout)[(size_t)mm * N + nn] = (half_t)val;
                else          ((float*)Cout)[(size_t)mm * N + nn]  = val;
            }
        }
}

// ---------------------------------------------------------------------------
// 4. Saliency conv
// ---------------------------------------------------------------------------
__global__ void conv_sal_kernel(const half_t* __restrict__ x1h,
                                const float* __restrict__ cw,
                                const float* __restrict__ cb,
                                float* __restrict__ sal) {
    const int idx = blockIdx.x * 256 + threadIdx.x;
    if (idx >= BH * SEQ) return;
    const int l  = idx & (SEQ - 1);
    const int bh = idx >> 11;
    const int h  = bh & (NHEADS - 1);
    const int b  = bh >> 3;
    const float* w = cw + (size_t)h * DCH * 3;
    float acc = cb[h];
#pragma unroll
    for (int dk = 0; dk < 3; ++dk) {
        const int ll = l + dk - 1;
        if (ll < 0 || ll >= SEQ) continue;
        const half_t* xr = x1h + ((size_t)b * SEQ + ll) * DCH;
        for (int c = 0; c < DCH; ++c)
            acc += (float)xr[c] * w[c * 3 + dk];
    }
    sal[idx] = acc;
}

// ---------------------------------------------------------------------------
// 5. V transpose: Vh (B,L,512) -> Vt (BH, 64, SEQ)  [K contiguous per d-row]
// ---------------------------------------------------------------------------
__global__ void vtrans_kernel(const half_t* __restrict__ Vh,
                              half_t* __restrict__ Vt) {
    const int idx = blockIdx.x * 256 + threadIdx.x;   // bh*64*2048 + d*2048 + k
    if (idx >= BH * DHEAD * SEQ) return;
    const int k  = idx & (SEQ - 1);
    const int d  = (idx >> 11) & (DHEAD - 1);
    const int bh = idx >> 17;
    const int h  = bh & (NHEADS - 1);
    const int b  = bh >> 3;
    Vt[idx] = Vh[((size_t)b * SEQ + k) * DCH + h * DHEAD + d];
}

// ---------------------------------------------------------------------------
// 6. FUSED attention: per (bh, 16-query tile)
//    Phase 0: async-stage sal row into LDS (ASYNCcnt path when available)
//    Phase 1: S = QK^T/8 + alpha*sal  -> LDS (16 x 2048 fp32, 128 KB)
//    Phase 2: softmax rows in LDS, single attn write to HBM
//    Phase 3: ctx = P @ V from LDS; B from transposed Vt (contiguous b128)
//    LDS: 131072 (S) + 1024 (red) + 4096 (part) + 8192 (sal) = 144384 B
// ---------------------------------------------------------------------------
__global__ void attn_fused_kernel(const half_t* __restrict__ Qh,
                                  const half_t* __restrict__ Kh,
                                  const half_t* __restrict__ Vt,
                                  const float* __restrict__ sal,
                                  const float* __restrict__ alpha_p,
                                  float* __restrict__ attn,
                                  half_t* __restrict__ ctxh) {
    extern __shared__ char smem[];
    float* Sf     = (float*)smem;            // 16 x 2048
    float* red    = Sf + 16 * SEQ;           // 256
    float* part   = red + 256;               // 4 x 256 split-K partials
    float* salLds = part + 1024;             // 2048

    const int qt = blockIdx.x;
    const int bh = blockIdx.y;
    const int b = bh >> 3, h = bh & 7;
    const int qBase = qt * 16;
    const half_t* Qb = Qh + (size_t)b * SEQ * DCH + h * DHEAD;
    const half_t* Kb = Kh + (size_t)b * SEQ * DCH + h * DHEAD;
    const half_t* Vtb = Vt + (size_t)bh * DHEAD * SEQ;

    const int wave = threadIdx.x >> 5;
    const int lane = threadIdx.x & 31;
    const int l16  = lane & 15;
    const int hi   = lane >> 4;
    const int kbA  = hi << 3;
    const int kbB  = hi << 3;
    const int mo   = hi * 8;
    const float alpha = alpha_p[0];

    // ---- Phase 0: stage sal row (8 KB) into LDS ----
    {
        const float* gsal = sal + (size_t)bh * SEQ;
#if HAVE_ASYNC_LDS
        for (int i = threadIdx.x; i < SEQ / 4; i += 256) {
            __builtin_amdgcn_global_load_async_to_lds_b128(
                (__attribute__((address_space(1))) vi4*)(gsal + i * 4),
                (__attribute__((address_space(3))) vi4*)(salLds + i * 4),
                0, 0);
        }
        __builtin_amdgcn_s_wait_asynccnt(0);
#else
        for (int i = threadIdx.x; i < SEQ; i += 256) salLds[i] = gsal[i];
#endif
    }
    __syncthreads();

    // ---- Phase 1: scores into LDS (each wave owns 256 keys) ----
    v16h aQ[2];
#pragma unroll
    for (int t = 0; t < 2; ++t)
        aQ[t] = load_a16x32(Qb + (size_t)(qBase + l16) * DCH + t * 32 + kbA);

    for (int nt = 0; nt < 16; ++nt) {
        const int keyBase = wave * 256 + nt * 16;
        v8f acc = {};
#pragma unroll
        for (int t = 0; t < 2; ++t) {
            v16h bK = load_b32x16_rowmajorN(
                Kb + (size_t)(keyBase + l16) * DCH + t * 32 + kbB);
            acc = wmma_f16(aQ[t], bK, acc);
        }
        const int nn = keyBase + l16;
        const float bias = alpha * salLds[nn];
#pragma unroll
        for (int r = 0; r < 8; ++r)
            Sf[(mo + r) * SEQ + nn] = acc[r] * 0.125f + bias;
    }
    __syncthreads();

    // ---- Phase 2: softmax (16 threads per row), single attn write ----
    {
        const int q   = threadIdx.x >> 4;
        const int t16 = threadIdx.x & 15;
        float* row = Sf + q * SEQ;

        float mx = -3.4e38f;
        for (int i = t16; i < SEQ; i += 16) mx = fmaxf(mx, row[i]);
        red[threadIdx.x] = mx;
        __syncthreads();
        if (t16 == 0) {
            float m = red[q * 16];
            for (int j = 1; j < 16; ++j) m = fmaxf(m, red[q * 16 + j]);
            red[q * 16] = m;
        }
        __syncthreads();
        mx = red[q * 16];

        float s = 0.f;
        for (int i = t16; i < SEQ; i += 16) {
            float e = __expf(row[i] - mx);
            row[i] = e; s += e;
        }
        __syncthreads();
        red[threadIdx.x] = s;
        __syncthreads();
        if (t16 == 0) {
            float t = 0.f;
            for (int j = 0; j < 16; ++j) t += red[q * 16 + j];
            red[q * 16] = t;
        }
        __syncthreads();
        const float inv = 1.f / red[q * 16];

        float* arow = attn + ((size_t)bh * SEQ + qBase + q) * SEQ;
        for (int i = t16; i < SEQ; i += 16) {
            const float p = row[i] * inv;
            row[i] = p;
            arow[i] = p;
        }
    }
    __syncthreads();

    // ---- Phase 3: ctx = P @ V from LDS; B tiles contiguous from Vt ----
    const int nt3   = wave & 3;          // d-tile 0..3 (N = 64)
    const int khalf = wave >> 2;         // K half 0/1
    const half_t* vtRow = Vtb + (size_t)(nt3 * 16 + l16) * SEQ;
    v8f accC = {};
    const int kEnd = khalf * 1024 + 1024;
    for (int k0 = khalf * 1024; k0 < kEnd; k0 += 32) {
        v16h aP;
        const float* ra = Sf + (size_t)l16 * SEQ + k0 + kbA;
#pragma unroll
        for (int v = 0; v < 8; ++v) {
            const int k = (v < 4) ? (2 * v) : (16 + 2 * (v - 4));
            aP[2 * v]     = (half_t)ra[k];
            aP[2 * v + 1] = (half_t)ra[k + 1];
        }
        v16h bV = load_b32x16_rowmajorN(vtRow + k0 + kbB);
        accC = wmma_f16(aP, bV, accC);
    }
    if (khalf == 1) {
#pragma unroll
        for (int r = 0; r < 8; ++r)
            part[nt3 * 256 + (mo + r) * 16 + l16] = accC[r];
    }
    __syncthreads();
    if (khalf == 0) {
#pragma unroll
        for (int r = 0; r < 8; ++r) {
            const float v = accC[r] + part[nt3 * 256 + (mo + r) * 16 + l16];
            ctxh[((size_t)b * SEQ + qBase + mo + r) * DCH + h * DHEAD + nt3 * 16 + l16] =
                (half_t)v;
        }
    }
}

// ---------------------------------------------------------------------------
// 7. Coupling: y2 = exp(clip(s,-2,2)) * x2 + t ; cat = [x1 | y2] in fp16
// ---------------------------------------------------------------------------
__global__ void couple_kernel(const float* __restrict__ sbuf,
                              const float* __restrict__ tbuf,
                              const float* __restrict__ x2f,
                              const half_t* __restrict__ x1h,
                              half_t* __restrict__ cath) {
    const size_t i = (size_t)blockIdx.x * 256 + threadIdx.x;
    if (i >= (size_t)ROWS * DCH) return;
    const size_t row = i / DCH;
    const int    c   = (int)(i % DCH);
    float ls = sbuf[i];
    ls = fminf(fmaxf(ls, -2.f), 2.f);
    const float y2 = __expf(ls) * x2f[i] + tbuf[i];
    cath[row * DMODEL + c]       = x1h[i];
    cath[row * DMODEL + DCH + c] = (half_t)y2;
}

// ---------------------------------------------------------------------------
// Launch
// ---------------------------------------------------------------------------
extern "C" void kernel_launch(void* const* d_in, const int* in_sizes, int n_in,
                              void* d_out, int out_size, void* d_ws, size_t ws_size,
                              hipStream_t stream) {
    (void)in_sizes; (void)n_in; (void)out_size; (void)ws_size;
    const float* src    = (const float*)d_in[0];
    const float* ln_g   = (const float*)d_in[1];
    const float* ln_b   = (const float*)d_in[2];
    const float* Wq     = (const float*)d_in[3];
    const float* bq     = (const float*)d_in[4];
    const float* Wk     = (const float*)d_in[5];
    const float* bk     = (const float*)d_in[6];
    const float* Wv     = (const float*)d_in[7];
    const float* bv     = (const float*)d_in[8];
    const float* conv_w = (const float*)d_in[9];
    const float* conv_b = (const float*)d_in[10];
    const float* alpha  = (const float*)d_in[11];
    const float* Ws     = (const float*)d_in[12];
    const float* bs     = (const float*)d_in[13];
    const float* Wt     = (const float*)d_in[14];
    const float* bt     = (const float*)d_in[15];
    const float* Wo     = (const float*)d_in[16];
    const float* bo     = (const float*)d_in[17];

    float* out_f  = (float*)d_out;                    // (B,L,1024)
    float* attn_f = out_f + (size_t)ROWS * DMODEL;    // (B,8,L,L)

    char* ws = (char*)d_ws;
    size_t off = 0;
    auto carve = [&](size_t bytes) { void* p = ws + off; off += (bytes + 255) & ~(size_t)255; return p; };
    half_t* x1h  = (half_t*)carve((size_t)ROWS * DCH * 2);
    float*  x2f  = (float*) carve((size_t)ROWS * DCH * 4);
    half_t* Wq_h = (half_t*)carve((size_t)DCH * DCH * 2);
    half_t* Wk_h = (half_t*)carve((size_t)DCH * DCH * 2);
    half_t* Wv_h = (half_t*)carve((size_t)DCH * DCH * 2);
    half_t* Ws_h = (half_t*)carve((size_t)DCH * DCH * 2);
    half_t* Wt_h = (half_t*)carve((size_t)DCH * DCH * 2);
    half_t* Wo_h = (half_t*)carve((size_t)DMODEL * DMODEL * 2);
    half_t* Qh   = (half_t*)carve((size_t)ROWS * DCH * 2);
    half_t* Kh   = (half_t*)carve((size_t)ROWS * DCH * 2);
    half_t* Vh   = (half_t*)carve((size_t)ROWS * DCH * 2);
    half_t* Vt   = (half_t*)carve((size_t)BH * DHEAD * SEQ * 2);
    float*  sal  = (float*) carve((size_t)BH * SEQ * 4);
    half_t* ctxh = (half_t*)carve((size_t)ROWS * DCH * 2);
    float*  sbuf = (float*) carve((size_t)ROWS * DCH * 4);
    float*  tbuf = (float*) carve((size_t)ROWS * DCH * 4);
    half_t* cath = (half_t*)carve((size_t)ROWS * DMODEL * 2);

    // 1. LayerNorm + split
    ln_split_kernel<<<ROWS, 256, 0, stream>>>(src, ln_g, ln_b, x1h, x2f);

    // 2. Weight conversions
    const int wsz = DCH * DCH;
    cvt_f32_f16_kernel<<<(wsz + 255) / 256, 256, 0, stream>>>(Wq, Wq_h, wsz);
    cvt_f32_f16_kernel<<<(wsz + 255) / 256, 256, 0, stream>>>(Wk, Wk_h, wsz);
    cvt_f32_f16_kernel<<<(wsz + 255) / 256, 256, 0, stream>>>(Wv, Wv_h, wsz);
    cvt_f32_f16_kernel<<<(wsz + 255) / 256, 256, 0, stream>>>(Ws, Ws_h, wsz);
    cvt_f32_f16_kernel<<<(wsz + 255) / 256, 256, 0, stream>>>(Wt, Wt_h, wsz);
    const int wosz = DMODEL * DMODEL;
    cvt_f32_f16_kernel<<<(wosz + 255) / 256, 256, 0, stream>>>(Wo, Wo_h, wosz);

    // 3. Q, K, V projections (M=8192, N=512, K=512)
    dim3 gProj(DCH / 64, ROWS / 128);
    gemm_wmma_kernel<true><<<gProj, 256, 0, stream>>>(x1h, Wq_h, bq, Qh, ROWS, DCH, DCH);
    gemm_wmma_kernel<true><<<gProj, 256, 0, stream>>>(x1h, Wk_h, bk, Kh, ROWS, DCH, DCH);
    gemm_wmma_kernel<true><<<gProj, 256, 0, stream>>>(x1h, Wv_h, bv, Vh, ROWS, DCH, DCH);

    // 4. Saliency conv + V transpose
    conv_sal_kernel<<<(BH * SEQ + 255) / 256, 256, 0, stream>>>(x1h, conv_w, conv_b, sal);
    vtrans_kernel<<<(BH * DHEAD * SEQ + 255) / 256, 256, 0, stream>>>(Vh, Vt);

    // 5. Fused scores + softmax + ctx (attn written exactly once)
    {
        const size_t shbytes = (size_t)16 * SEQ * 4 + 256 * 4 + 1024 * 4 + SEQ * 4;
        dim3 gAtt(SEQ / 16, BH);
        attn_fused_kernel<<<gAtt, 256, shbytes, stream>>>(Qh, Kh, Vt, sal, alpha,
                                                          attn_f, ctxh);
    }

    // 6. scale / shift projections (fp32 out)
    gemm_wmma_kernel<false><<<gProj, 256, 0, stream>>>(ctxh, Ws_h, bs, sbuf, ROWS, DCH, DCH);
    gemm_wmma_kernel<false><<<gProj, 256, 0, stream>>>(ctxh, Wt_h, bt, tbuf, ROWS, DCH, DCH);

    // 7. coupling + concat
    const size_t nCouple = (size_t)ROWS * DCH;
    couple_kernel<<<(unsigned)((nCouple + 255) / 256), 256, 0, stream>>>(sbuf, tbuf, x2f, x1h, cath);

    // 8. final projection (M=8192, N=1024, K=1024) -> out
    dim3 gOut(DMODEL / 64, ROWS / 128);
    gemm_wmma_kernel<false><<<gOut, 256, 0, stream>>>(cath, Wo_h, bo, out_f, ROWS, DMODEL, DMODEL);
}
